// GraphGAT_88072599372183
// MI455X (gfx1250) — matrix operand order repaired and checked by reading, not compiled, not run on hardware
//
#include <hip/hip_runtime.h>
#include <hip/hip_bf16.h>

#define N_NODES    50000
#define N_EDGES    400000
#define ETOT       (N_EDGES + N_NODES)
#define F_IN       128
#define HID        64
#define HEADS      8
#define HC1        (HEADS * HID)   /* 512 */
#define NUM_GRAPHS 64
#define NUM_CLASSES 2
#define NEG_SLOPE  0.2f

typedef __attribute__((ext_vector_type(16))) __bf16         v16bf;
typedef __attribute__((ext_vector_type(8)))  float          v8f;
typedef __attribute__((ext_vector_type(8)))  unsigned int   v8ui;
typedef __attribute__((ext_vector_type(4)))  unsigned int   v4ui;

#define CDIV(a, b) (((a) + (b) - 1) / (b))

__device__ __forceinline__ unsigned short f2bf(float f) {
  unsigned int u = __float_as_uint(f);
  u += 0x7FFFu + ((u >> 16) & 1u);        // round-to-nearest-even
  return (unsigned short)(u >> 16);
}
__device__ __forceinline__ float bf2f(unsigned short s) {
  return __uint_as_float(((unsigned int)s) << 16);
}
__device__ __forceinline__ void atomicMaxF(float* addr, float val) {
  unsigned int* ua = (unsigned int*)addr;
  unsigned int cur = *ua;
  while (true) {
    if (__uint_as_float(cur) >= val) break;
    unsigned int assumed = cur;
    cur = atomicCAS(ua, assumed, __float_as_uint(val));
    if (cur == assumed) break;
  }
}

// ---------------------------------------------------------------------------
// f32 -> bf16 conversion / fill
// ---------------------------------------------------------------------------
__global__ void k_f2bf(const float* __restrict__ in, unsigned short* __restrict__ out,
                       size_t n) {
  size_t i = (size_t)blockIdx.x * blockDim.x + threadIdx.x;
  if (i < n) out[i] = f2bf(in[i]);
}

__global__ void k_fill(float* __restrict__ p, float v, size_t n) {
  size_t i = (size_t)blockIdx.x * blockDim.x + threadIdx.x;
  if (i < n) p[i] = v;
}

// ---------------------------------------------------------------------------
// Pack weight matrix W[K,Ncol] (f32) into WMMA B-fragment-native bf16 layout:
//   Bp[ ((tile_n * KC + kc) * 32 + lane) * 16 + j ]
// where for element (k,n): kc=k/32, half=((k%32)/8)&1, j=(k%8)+8*((k%32)/16),
//   lane=half*16 + (n%16), tile_n=n/16.
// Each wave's per-k-chunk B fragment becomes 32 contiguous bytes per lane,
// lane-consecutive (coalesced b128 pairs, no per-element address math).
// ---------------------------------------------------------------------------
__global__ void k_packB(const float* __restrict__ W, unsigned short* __restrict__ Bp,
                        int K, int Ncol) {
  int i = blockIdx.x * blockDim.x + threadIdx.x;
  if (i >= K * Ncol) return;
  int k = i / Ncol, n = i - k * Ncol;
  int kc = k >> 5;
  int off = k & 31;
  int half = (off >> 3) & 1;
  int j = (off & 7) | ((off >> 4) << 3);
  int tile = n >> 4;
  int lane = (half << 4) | (n & 15);
  int KC = K >> 5;
  size_t dst = (((size_t)tile * KC + kc) * 32 + lane) * 16 + j;
  Bp[dst] = f2bf(W[i]);
}

// ---------------------------------------------------------------------------
// WMMA GEMM: C_bf16[M,64*NG-strip] = A_bf16[M,K] @ Bpacked, f32 accumulate.
// One wave -> 16 rows x 64 cols (4 accumulators); K-loop fully unrolled via
// template. A fragment: two b128 loads (ISA 16-bit A 16x32 layout). B
// fragment: one 32B contiguous load per tile from the packed layout.
// Wave-uniform exit keeps EXEC all-ones for WMMA.
// ---------------------------------------------------------------------------
template <int KCHUNKS>
__global__ void k_gemm_bf16(const unsigned short* __restrict__ A,
                            const unsigned short* __restrict__ Bp,
                            unsigned short* __restrict__ Cout,
                            int M, int Ncol) {
  const int K = KCHUNKS * 32;
  int wid = blockIdx.x * (blockDim.x >> 5) + (threadIdx.x >> 5);
  int ng = Ncol >> 6;                      // groups of four 16-col tiles
  int tm = wid / ng;
  int tg = wid - tm * ng;
  if (tm >= (M >> 4)) return;              // wave-uniform

  int lane = threadIdx.x & 31;
  int half = lane >> 4;
  int l15  = lane & 15;

  const unsigned short* arow = A + (size_t)(tm * 16 + l15) * K + half * 8;
  // lane's slice of packed B for the 4 tiles of this group
  const unsigned short* bb = Bp + ((size_t)(tg * 4) * KCHUNKS * 32 + lane) * 16;

  v8f acc0 = {}, acc1 = {}, acc2 = {}, acc3 = {};
#pragma unroll
  for (int kc = 0; kc < KCHUNKS; ++kc) {
    int k0 = kc << 5;
    __builtin_prefetch(arow + k0 + 32, 0, 3);       // global_prefetch_b8
    // A: j=0..7 -> K=k0+half*8+j ; j=8..15 -> K=k0+half*8+j+8
    v4ui alo = *(const v4ui*)(arow + k0);
    v4ui ahi = *(const v4ui*)(arow + k0 + 16);
    v8ui apk;
#pragma unroll
    for (int j = 0; j < 4; ++j) { apk[j] = alo[j]; apk[j + 4] = ahi[j]; }
    v16bf av = __builtin_bit_cast(v16bf, apk);

    const unsigned short* bt = bb + (size_t)kc * 32 * 16;
    const size_t tstride = (size_t)KCHUNKS * 32 * 16;
    v16bf bv0 = __builtin_bit_cast(v16bf, *(const v8ui*)(bt));
    v16bf bv1 = __builtin_bit_cast(v16bf, *(const v8ui*)(bt + tstride));
    v16bf bv2 = __builtin_bit_cast(v16bf, *(const v8ui*)(bt + 2 * tstride));
    v16bf bv3 = __builtin_bit_cast(v16bf, *(const v8ui*)(bt + 3 * tstride));

    acc0 = __builtin_amdgcn_wmma_f32_16x16x32_bf16(false, av, false, bv0, (short)0, acc0, false, false);
    acc1 = __builtin_amdgcn_wmma_f32_16x16x32_bf16(false, av, false, bv1, (short)0, acc1, false, false);
    acc2 = __builtin_amdgcn_wmma_f32_16x16x32_bf16(false, av, false, bv2, (short)0, acc2, false, false);
    acc3 = __builtin_amdgcn_wmma_f32_16x16x32_bf16(false, av, false, bv3, (short)0, acc3, false, false);
  }

  // C/D layout: VGPR r -> row tm*16 + half*8 + r, col = tile_base + l15
  unsigned short* cp = Cout + (size_t)(tm * 16 + half * 8) * Ncol + tg * 64 + l15;
#pragma unroll
  for (int r = 0; r < 8; ++r) {
    size_t ro = (size_t)r * Ncol;
    cp[ro]      = f2bf(acc0[r]);
    cp[ro + 16] = f2bf(acc1[r]);
    cp[ro + 32] = f2bf(acc2[r]);
    cp[ro + 48] = f2bf(acc3[r]);
  }
}

// ---------------------------------------------------------------------------
// Attention scores: a_s[n,h] = <h[n,h,:], att_src[h,:]>, same for a_d
// ---------------------------------------------------------------------------
__global__ void k_att(const unsigned short* __restrict__ h,
                      const float* __restrict__ asrc, const float* __restrict__ adst,
                      float* __restrict__ as_, float* __restrict__ ad_,
                      int n_nodes, int H, int C) {
  int idx = blockIdx.x * blockDim.x + threadIdx.x;
  if (idx >= n_nodes * H) return;
  int n = idx / H, hh = idx - n * H;
  const unsigned short* hp = h + (size_t)n * H * C + hh * C;
  const float* s = asrc + hh * C;
  const float* d = adst + hh * C;
  float ss = 0.f, dd = 0.f;
  for (int c = 0; c < C; ++c) {
    float v = bf2f(hp[c]);
    ss += v * s[c];
    dd += v * d[c];
  }
  as_[idx] = ss;
  ad_[idx] = dd;
}

// ---------------------------------------------------------------------------
// Edge kernels (edge id e < N_EDGES from edge_index; else self-loop e-N_EDGES)
// ---------------------------------------------------------------------------
__global__ void k_edge_max(const int* __restrict__ esrc, const int* __restrict__ edst,
                           const float* __restrict__ as_, const float* __restrict__ ad_,
                           float* __restrict__ mx, int H) {
  int idx = blockIdx.x * blockDim.x + threadIdx.x;
  if (idx >= ETOT * H) return;
  int e = idx / H, hh = idx - e * H;
  int s = (e < N_EDGES) ? esrc[e] : (e - N_EDGES);
  int d = (e < N_EDGES) ? edst[e] : (e - N_EDGES);
  float v = as_[s * H + hh] + ad_[d * H + hh];
  v = v > 0.f ? v : v * NEG_SLOPE;
  atomicMaxF(&mx[d * H + hh], v);
}

__global__ void k_edge_expsum(const int* __restrict__ esrc, const int* __restrict__ edst,
                              const float* __restrict__ as_, const float* __restrict__ ad_,
                              const float* __restrict__ mx,
                              float* __restrict__ ew, float* __restrict__ dn, int H) {
  int idx = blockIdx.x * blockDim.x + threadIdx.x;
  if (idx >= ETOT * H) return;
  int e = idx / H, hh = idx - e * H;
  int s = (e < N_EDGES) ? esrc[e] : (e - N_EDGES);
  int d = (e < N_EDGES) ? edst[e] : (e - N_EDGES);
  float v = as_[s * H + hh] + ad_[d * H + hh];
  v = v > 0.f ? v : v * NEG_SLOPE;
  float w = __expf(v - mx[d * H + hh]);
  ew[idx] = w;
  atomicAdd(&dn[d * H + hh], w);
}

__global__ void k_edge_aggr(const int* __restrict__ esrc, const int* __restrict__ edst,
                            const unsigned short* __restrict__ h,
                            const float* __restrict__ ew, const float* __restrict__ dn,
                            float* __restrict__ out, int H, int C) {
  long long idx = (long long)blockIdx.x * blockDim.x + threadIdx.x;
  int HCl = H * C;
  if (idx >= (long long)ETOT * HCl) return;
  int e = (int)(idx / HCl);
  int c = (int)(idx - (long long)e * HCl);
  int hh = c / C;
  int s = (e < N_EDGES) ? esrc[e] : (e - N_EDGES);
  int d = (e < N_EDGES) ? edst[e] : (e - N_EDGES);
  float coef = ew[(size_t)e * H + hh] / dn[(size_t)d * H + hh];
  atomicAdd(&out[(size_t)d * HCl + c], bf2f(h[(size_t)s * HCl + c]) * coef);
}

// ---------------------------------------------------------------------------
// Bias + ELU epilogues
// ---------------------------------------------------------------------------
__global__ void k_bias_elu_bf16(const float* __restrict__ acc, const float* __restrict__ b,
                                unsigned short* __restrict__ ob, long long total, int C) {
  long long i = (long long)blockIdx.x * blockDim.x + threadIdx.x;
  if (i >= total) return;
  int c = (int)(i % C);
  float v = acc[i] + b[c];
  v = v > 0.f ? v : (__expf(v) - 1.f);
  ob[i] = f2bf(v);
}

__global__ void k_bias_elu_f32(const float* __restrict__ acc, const float* __restrict__ b,
                               float* __restrict__ of, long long total, int C) {
  long long i = (long long)blockIdx.x * blockDim.x + threadIdx.x;
  if (i >= total) return;
  int c = (int)(i % C);
  float v = acc[i] + b[c];
  of[i] = v > 0.f ? v : (__expf(v) - 1.f);
}

// ---------------------------------------------------------------------------
// Global mean pool + final FC
// ---------------------------------------------------------------------------
__global__ void k_pool(const float* __restrict__ hf, const int* __restrict__ batch,
                       float* __restrict__ sums, float* __restrict__ cnt) {
  int idx = blockIdx.x * blockDim.x + threadIdx.x;
  if (idx >= N_NODES * HID) return;
  int n = idx / HID, c = idx - n * HID;
  int g = batch[n];
  atomicAdd(&sums[g * HID + c], hf[idx]);
  if (c == 0) atomicAdd(&cnt[g], 1.0f);
}

__global__ void k_final(const float* __restrict__ sums, const float* __restrict__ cnt,
                        const float* __restrict__ fcw, const float* __restrict__ fcb,
                        float* __restrict__ out) {
  int idx = blockIdx.x * blockDim.x + threadIdx.x;
  if (idx >= NUM_GRAPHS * NUM_CLASSES) return;
  int g = idx / NUM_CLASSES, k = idx - g * NUM_CLASSES;
  float c = cnt[g];
  c = c > 1.f ? c : 1.f;
  float a = fcb[k];
  for (int j = 0; j < HID; ++j)
    a += (sums[g * HID + j] / c) * fcw[j * NUM_CLASSES + k];
  out[idx] = a;
}

// ---------------------------------------------------------------------------
extern "C" void kernel_launch(void* const* d_in, const int* in_sizes, int n_in,
                              void* d_out, int out_size, void* d_ws, size_t ws_size,
                              hipStream_t stream) {
  const float* x        = (const float*)d_in[0];
  const int*   ei       = (const int*)d_in[1];   // [2, E] flat
  const int*   batch    = (const int*)d_in[2];
  const float* W1       = (const float*)d_in[3];
  const float* att_src1 = (const float*)d_in[4];
  const float* att_dst1 = (const float*)d_in[5];
  const float* b1       = (const float*)d_in[6];
  const float* W2       = (const float*)d_in[7];
  const float* att_src2 = (const float*)d_in[8];
  const float* att_dst2 = (const float*)d_in[9];
  const float* b2       = (const float*)d_in[10];
  const float* fc_w     = (const float*)d_in[11];
  const float* fc_b     = (const float*)d_in[12];
  const int* esrc = ei;
  const int* edst = ei + N_EDGES;

  (void)in_sizes; (void)n_in; (void)out_size; (void)ws_size;

  // ---- workspace carve (256B aligned) ----
  char* wsp = (char*)d_ws;
  auto carve = [&](size_t bytes) -> void* {
    void* p = (void*)wsp;
    wsp += (bytes + 255) & ~(size_t)255;
    return p;
  };
  unsigned short* xb  = (unsigned short*)carve((size_t)N_NODES * F_IN * 2);
  unsigned short* w1p = (unsigned short*)carve((size_t)F_IN * HC1 * 2);   // packed B1
  unsigned short* w2p = (unsigned short*)carve((size_t)HC1 * HID * 2);    // packed B2
  unsigned short* h1b = (unsigned short*)carve((size_t)N_NODES * HC1 * 2); // reused as x2 bf16
  float* as1  = (float*)carve((size_t)N_NODES * HEADS * 4);
  float* ad1  = (float*)carve((size_t)N_NODES * HEADS * 4);
  float* mx1  = (float*)carve((size_t)N_NODES * HEADS * 4);
  float* dn1  = (float*)carve((size_t)N_NODES * HEADS * 4);
  float* ew1  = (float*)carve((size_t)ETOT * HEADS * 4);
  float* out1 = (float*)carve((size_t)N_NODES * HC1 * 4);
  unsigned short* h2b = (unsigned short*)carve((size_t)N_NODES * HID * 2);
  float* as2  = (float*)carve((size_t)N_NODES * 4);
  float* ad2  = (float*)carve((size_t)N_NODES * 4);
  float* mx2  = (float*)carve((size_t)N_NODES * 4);
  float* dn2  = (float*)carve((size_t)N_NODES * 4);
  float* ew2  = (float*)carve((size_t)ETOT * 4);
  float* out2 = (float*)carve((size_t)N_NODES * HID * 4);
  float* hf   = (float*)carve((size_t)N_NODES * HID * 4);
  float* sums = (float*)carve((size_t)NUM_GRAPHS * HID * 4);
  float* cnt  = (float*)carve((size_t)NUM_GRAPHS * 4);

  const int T = 256;

  // ---- bf16 conversion of activations + weight packing ----
  k_f2bf<<<CDIV((size_t)N_NODES * F_IN, T), T, 0, stream>>>(x, xb, (size_t)N_NODES * F_IN);
  k_packB<<<CDIV(F_IN * HC1, T), T, 0, stream>>>(W1, w1p, F_IN, HC1);
  k_packB<<<CDIV(HC1 * HID, T), T, 0, stream>>>(W2, w2p, HC1, HID);

  // ---- layer 1: h1 = x @ W1 (WMMA bf16, f32 acc), 16x64 strip per wave ----
  {
    int waves = (N_NODES / 16) * (HC1 / 64);
    k_gemm_bf16<F_IN / 32><<<CDIV(waves, 8), T, 0, stream>>>(xb, w1p, h1b, N_NODES, HC1);
  }
  k_att<<<CDIV(N_NODES * HEADS, T), T, 0, stream>>>(h1b, att_src1, att_dst1, as1, ad1,
                                                    N_NODES, HEADS, HID);
  k_fill<<<CDIV((size_t)N_NODES * HEADS, T), T, 0, stream>>>(mx1, -1e30f, (size_t)N_NODES * HEADS);
  k_fill<<<CDIV((size_t)N_NODES * HEADS, T), T, 0, stream>>>(dn1, 0.f, (size_t)N_NODES * HEADS);
  k_fill<<<CDIV((size_t)N_NODES * HC1, T), T, 0, stream>>>(out1, 0.f, (size_t)N_NODES * HC1);
  k_edge_max<<<CDIV(ETOT * HEADS, T), T, 0, stream>>>(esrc, edst, as1, ad1, mx1, HEADS);
  k_edge_expsum<<<CDIV(ETOT * HEADS, T), T, 0, stream>>>(esrc, edst, as1, ad1, mx1, ew1, dn1, HEADS);
  {
    long long tot = (long long)ETOT * HC1;
    k_edge_aggr<<<(unsigned)((tot + T - 1) / T), T, 0, stream>>>(esrc, edst, h1b, ew1, dn1,
                                                                 out1, HEADS, HID);
  }
  // x2 = elu(out1 + b1) -> bf16, overwriting h1b (safe: h1b no longer needed)
  {
    long long tot = (long long)N_NODES * HC1;
    k_bias_elu_bf16<<<(unsigned)((tot + T - 1) / T), T, 0, stream>>>(out1, b1, h1b, tot, HC1);
  }

  // ---- layer 2: h2 = x2 @ W2 (WMMA) ----
  {
    int waves = (N_NODES / 16) * (HID / 64);
    k_gemm_bf16<HC1 / 32><<<CDIV(waves, 8), T, 0, stream>>>(h1b, w2p, h2b, N_NODES, HID);
  }
  k_att<<<CDIV(N_NODES, T), T, 0, stream>>>(h2b, att_src2, att_dst2, as2, ad2, N_NODES, 1, HID);
  k_fill<<<CDIV((size_t)N_NODES, T), T, 0, stream>>>(mx2, -1e30f, (size_t)N_NODES);
  k_fill<<<CDIV((size_t)N_NODES, T), T, 0, stream>>>(dn2, 0.f, (size_t)N_NODES);
  k_fill<<<CDIV((size_t)N_NODES * HID, T), T, 0, stream>>>(out2, 0.f, (size_t)N_NODES * HID);
  k_fill<<<CDIV((size_t)NUM_GRAPHS * HID, T), T, 0, stream>>>(sums, 0.f, (size_t)NUM_GRAPHS * HID);
  k_fill<<<1, T, 0, stream>>>(cnt, 0.f, (size_t)NUM_GRAPHS);
  k_edge_max<<<CDIV(ETOT, T), T, 0, stream>>>(esrc, edst, as2, ad2, mx2, 1);
  k_edge_expsum<<<CDIV(ETOT, T), T, 0, stream>>>(esrc, edst, as2, ad2, mx2, ew2, dn2, 1);
  {
    long long tot = (long long)ETOT * HID;
    k_edge_aggr<<<(unsigned)((tot + T - 1) / T), T, 0, stream>>>(esrc, edst, h2b, ew2, dn2,
                                                                 out2, 1, HID);
  }
  {
    long long tot = (long long)N_NODES * HID;
    k_bias_elu_f32<<<(unsigned)((tot + T - 1) / T), T, 0, stream>>>(out2, b2, hf, tot, HID);
  }

  // ---- pool + FC ----
  k_pool<<<CDIV(N_NODES * HID, T), T, 0, stream>>>(hf, batch, sums, cnt);
  k_final<<<1, 128, 0, stream>>>(sums, cnt, fc_w, fc_b, (float*)d_out);
}
